// Net_21500606283860
// MI455X (gfx1250) — matrix operand order repaired and checked by reading, not compiled, not run on hardware
//
#include <hip/hip_runtime.h>
#include <hip/hip_bf16.h>

#define N_NODES 100000
#define IN_DIM  1433
#define HID_DIM 16
#define OUT_DIM 7
#define N_EDGES 3200000

#define KCHUNKS 45                      // ceil(1433/32)
#define LDSB_ELEMS (KCHUNKS * 32 * 16)  // 23040 bf16 = 46 KB LDS

typedef __attribute__((ext_vector_type(16))) __bf16 v16bf;
typedef __attribute__((ext_vector_type(8)))  float  v8f;
// 4-dword vector with 4-byte alignment: rows of x are only 4B-aligned
// (1433 floats pitch). gfx1250 handles unaligned b128 loads natively.
typedef __attribute__((ext_vector_type(4), aligned(4))) float f32x4u;

// ---------------- degree kernels ----------------
__global__ __launch_bounds__(256) void k_deg_init(float* __restrict__ deg) {
    int i = blockIdx.x * 256 + threadIdx.x;
    if (i < N_NODES) deg[i] = 1.0f;   // self loop
}

__global__ __launch_bounds__(256) void k_deg_count(const long long* __restrict__ ei,
                                                   float* __restrict__ deg) {
    int e = blockIdx.x * 256 + threadIdx.x;
    if (e < N_EDGES) atomicAdd(&deg[(int)ei[e]], 1.0f);
}

__global__ __launch_bounds__(256) void k_deg_rsqrt(float* __restrict__ deg) {
    int i = blockIdx.x * 256 + threadIdx.x;
    if (i < N_NODES) deg[i] = rsqrtf(deg[i]);   // in place: deg -> d^-1/2
}

// ---------------- layer-1 GEMM: hs = dinv * (x @ W1 + b1), agg1 init = hs ---------------
__global__ __launch_bounds__(256)
void k_gemm1(const float* __restrict__ x, const float* __restrict__ W1,
             const float* __restrict__ b1, const float* __restrict__ dinv,
             float* __restrict__ hs, float* __restrict__ agg1) {
    // W1 staged as bf16 in B-fragment order: g = c*512 + lane*16 + j
    // B layout (16-bit, 32x16): lane = N, K packed 2/VGPR; lanes 0-15 K 0-7/16-23,
    // lanes 16-31 K 8-15/24-31.
    __shared__ __attribute__((aligned(32))) __bf16 ldsB[LDSB_ELEMS];
    for (int g = threadIdx.x; g < LDSB_ELEMS; g += 256) {
        int c = g >> 9;
        int w = g & 511;
        int l = w >> 4;
        int j = w & 15;
        int k = 32 * c + ((l >= 16) ? 8 : 0) + ((j >= 8) ? 8 : 0) + j;
        int n = l & 15;
        float v = (k < IN_DIM) ? W1[k * HID_DIM + n] : 0.0f;
        ldsB[g] = (__bf16)v;
    }
    __syncthreads();

    int lane = threadIdx.x & 31;
    int wave = threadIdx.x >> 5;
    int tile = blockIdx.x * 8 + wave;
    if (tile >= N_NODES / 16) return;     // 6250 tiles exactly

    int row0 = tile * 16;
    int m    = lane & 15;                 // A-matrix row for this lane
    int row  = row0 + m;
    int base = (lane >= 16) ? 8 : 0;      // A: lanes 0-15 K 0-7/16-23, lanes 16-31 K 8-15/24-31
    const float* xrow = x + (long)row * IN_DIM + base;

    v8f acc = {};
    for (int c = 0; c < KCHUNKS - 1; ++c) {        // full chunks: K 0..1407
        int k0 = c * 32;
        __builtin_prefetch(xrow + k0 + 160, 0, 1); // stream-ahead hint
        // x is read exactly once (573 MB): non-temporal so it does not evict
        // the L2-resident edge_index / hs / agg buffers reused by later kernels.
        f32x4u p0 = __builtin_nontemporal_load((const f32x4u*)(xrow + k0));
        f32x4u p1 = __builtin_nontemporal_load((const f32x4u*)(xrow + k0 + 4));
        f32x4u p2 = __builtin_nontemporal_load((const f32x4u*)(xrow + k0 + 16));
        f32x4u p3 = __builtin_nontemporal_load((const f32x4u*)(xrow + k0 + 20));
        v16bf a;
#pragma unroll
        for (int j = 0; j < 4; ++j) {
            a[j]      = (__bf16)p0[j];
            a[4 + j]  = (__bf16)p1[j];
            a[8 + j]  = (__bf16)p2[j];
            a[12 + j] = (__bf16)p3[j];
        }
        v16bf b = *reinterpret_cast<const v16bf*>(&ldsB[(c << 9) + (lane << 4)]);
        acc = __builtin_amdgcn_wmma_f32_16x16x32_bf16(false, a, false, b,
                                                      (short)0, acc, false, false);
    }
    {   // tail chunk c = 44: K 1408..1439, valid K < 1433.
        // Group 1 (a[0..7], K = 1408+base..1415+base <= 1423): always valid.
        // Group 2 (a[8..15], K = 1424+base+j): a[8] always valid (<=1432);
        // a[9..15] valid only for lanes 0-15 (base == 0).
        int c = KCHUNKS - 1, k0 = c * 32;
        bool lo = (lane < 16);
        v16bf a;
#pragma unroll
        for (int j = 0; j < 8; ++j) a[j] = (__bf16)xrow[k0 + j];
        a[8] = (__bf16)xrow[k0 + 16];
#pragma unroll
        for (int j = 1; j < 8; ++j) {
            int kj = lo ? (k0 + 16 + j) : 0;       // clamp: no OOB read
            float fv = xrow[kj];
            a[8 + j] = lo ? (__bf16)fv : (__bf16)0.0f;
        }
        v16bf b = *reinterpret_cast<const v16bf*>(&ldsB[(c << 9) + (lane << 4)]);
        acc = __builtin_amdgcn_wmma_f32_16x16x32_bf16(false, a, false, b,
                                                      (short)0, acc, false, false);
    }

    // C/D layout: vgpr v, lane l -> row = row0 + v + (l>=16 ? 8:0), col = l&15
    int mo  = (lane >= 16) ? 8 : 0;
    int col = lane & 15;
    float bias = b1[col];
#pragma unroll
    for (int v = 0; v < 8; ++v) {
        int r = row0 + mo + v;
        float d   = dinv[r];
        float val = d * (acc[v] + bias);
        hs[r * HID_DIM + col]   = val;   // message payload
        agg1[r * HID_DIM + col] = val;   // self-loop init of aggregation
    }
}

// ---------------- layer-1 edge scatter: agg1[dst] += hs[src] (4 thr/edge) --------------
__global__ __launch_bounds__(256)
void k_agg1(const long long* __restrict__ ei, const float* __restrict__ hs,
            float* __restrict__ agg1) {
    long long gid = (long long)blockIdx.x * 256 + threadIdx.x;
    if (gid >= (long long)N_EDGES * 4) return;
    int e  = (int)(gid >> 2);
    int f0 = ((int)gid & 3) * 4;
    int s  = (int)ei[e];
    int d  = (int)ei[N_EDGES + e];
    const float4 v = *reinterpret_cast<const float4*>(&hs[(long)s * HID_DIM + f0]);
    float* dst = &agg1[(long)d * HID_DIM + f0];
    atomicAdd(dst + 0, v.x); atomicAdd(dst + 1, v.y);
    atomicAdd(dst + 2, v.z); atomicAdd(dst + 3, v.w);
}

// ---- mid: z = relu(dinv*agg1); hs2 = dinv*(z@W2 + b2); agg2 init = hs2 (pad to 8) ----
__global__ __launch_bounds__(256)
void k_mid(const float* __restrict__ agg1, const float* __restrict__ dinv,
           const float* __restrict__ W2, const float* __restrict__ b2,
           float* __restrict__ hs2, float* __restrict__ agg2) {
    __shared__ float w[16 * 7 + 7];
    if (threadIdx.x < 119)
        w[threadIdx.x] = (threadIdx.x < 112) ? W2[threadIdx.x] : b2[threadIdx.x - 112];
    __syncthreads();
    int i = blockIdx.x * 256 + threadIdx.x;
    if (i >= N_NODES) return;
    float d = dinv[i];
    float z[16];
#pragma unroll
    for (int k = 0; k < 16; ++k) z[k] = fmaxf(d * agg1[(long)i * 16 + k], 0.0f);
#pragma unroll
    for (int j = 0; j < 7; ++j) {
        float h = w[112 + j];
#pragma unroll
        for (int k = 0; k < 16; ++k) h += z[k] * w[k * 7 + j];
        float val = d * h;
        hs2[(long)i * 8 + j]  = val;
        agg2[(long)i * 8 + j] = val;
    }
    hs2[(long)i * 8 + 7]  = 0.0f;
    agg2[(long)i * 8 + 7] = 0.0f;
}

// ---------------- layer-2 edge scatter (2 thr/edge, feature 7 is padding) -------------
__global__ __launch_bounds__(256)
void k_agg2(const long long* __restrict__ ei, const float* __restrict__ hs2,
            float* __restrict__ agg2) {
    long long gid = (long long)blockIdx.x * 256 + threadIdx.x;
    if (gid >= (long long)N_EDGES * 2) return;
    int e  = (int)(gid >> 1);
    int f0 = ((int)gid & 1) * 4;
    int s  = (int)ei[e];
    int d  = (int)ei[N_EDGES + e];
    const float4 v = *reinterpret_cast<const float4*>(&hs2[(long)s * 8 + f0]);
    float* dst = &agg2[(long)d * 8 + f0];
    atomicAdd(dst + 0, v.x); atomicAdd(dst + 1, v.y); atomicAdd(dst + 2, v.z);
    if (f0 == 0) atomicAdd(dst + 3, v.w);   // skip pad feature 7
}

// ---------------- final: out = log_softmax(dinv * agg2) -------------------------------
__global__ __launch_bounds__(256)
void k_out(const float* __restrict__ agg2, const float* __restrict__ dinv,
           float* __restrict__ out) {
    int i = blockIdx.x * 256 + threadIdx.x;
    if (i >= N_NODES) return;
    float d = dinv[i];
    float v[7], m = -INFINITY;
#pragma unroll
    for (int j = 0; j < 7; ++j) { v[j] = d * agg2[(long)i * 8 + j]; m = fmaxf(m, v[j]); }
    float s = 0.0f;
#pragma unroll
    for (int j = 0; j < 7; ++j) s += __expf(v[j] - m);
    float ls = __logf(s);
#pragma unroll
    for (int j = 0; j < 7; ++j) out[(long)i * 7 + j] = v[j] - m - ls;
}

static inline size_t alignup(size_t v) { return (v + 255) & ~(size_t)255; }

extern "C" void kernel_launch(void* const* d_in, const int* in_sizes, int n_in,
                              void* d_out, int out_size, void* d_ws, size_t ws_size,
                              hipStream_t stream) {
    const float*     x  = (const float*)d_in[0];
    const long long* ei = (const long long*)d_in[1];   // int64 per reference
    const float*     W1 = (const float*)d_in[2];
    const float*     b1 = (const float*)d_in[3];
    const float*     W2 = (const float*)d_in[4];
    const float*     b2 = (const float*)d_in[5];
    float* out = (float*)d_out;

    char* ws = (char*)d_ws;
    size_t off = 0;
    float* dinv = (float*)(ws + off); off += alignup((size_t)N_NODES * 4);
    float* hs   = (float*)(ws + off); off += alignup((size_t)N_NODES * 16 * 4);
    float* agg1 = (float*)(ws + off); off += alignup((size_t)N_NODES * 16 * 4);
    float* hs2  = (float*)(ws + off); off += alignup((size_t)N_NODES * 8 * 4);
    float* agg2 = (float*)(ws + off); off += alignup((size_t)N_NODES * 8 * 4);

    const int nb_nodes = (N_NODES + 255) / 256;
    const int nb_edges = (N_EDGES + 255) / 256;

    k_deg_init <<<nb_nodes, 256, 0, stream>>>(dinv);
    k_deg_count<<<nb_edges, 256, 0, stream>>>(ei, dinv);
    k_deg_rsqrt<<<nb_nodes, 256, 0, stream>>>(dinv);

    const int n_tiles = N_NODES / 16;                 // 6250
    k_gemm1<<<(n_tiles + 7) / 8, 256, 0, stream>>>(x, W1, b1, dinv, hs, agg1);

    k_agg1<<<(int)(((long long)N_EDGES * 4 + 255) / 256), 256, 0, stream>>>(ei, hs, agg1);
    k_mid <<<nb_nodes, 256, 0, stream>>>(agg1, dinv, W2, b2, hs2, agg2);
    k_agg2<<<(int)(((long long)N_EDGES * 2 + 255) / 256), 256, 0, stream>>>(ei, hs2, agg2);
    k_out <<<nb_nodes, 256, 0, stream>>>(agg2, dinv, out);
}
